// StandardROIHeads_62337155334367
// MI455X (gfx1250) — compile-verified, matrix operand order checked by reading, not run on hardware
//
#include <hip/hip_runtime.h>
#include <hip/hip_bf16.h>

// ---------------------------------------------------------------------------
// StandardROIHeads pipeline for MI455X (gfx1250, wave32, WMMA).
//  - sampling / matching : tiny, one block per image
//  - ROIAlign            : NHWC features, coalesced, bin-major pooled layout
//  - FC1/FC2/heads       : f16 WMMA GEMM, split-K=4, double-buffered LDS,
//                          async global->LDS when the toolchain exposes it.
// Workspace: ~84 MB carved from d_ws.
// ---------------------------------------------------------------------------

typedef _Float16 h16_t;
typedef _Float16 half8 __attribute__((ext_vector_type(8)));
typedef _Float16 v16h  __attribute__((ext_vector_type(16)));
typedef float    v8f   __attribute__((ext_vector_type(8)));

#define NP 2000     // proposals per image
#define NG 64       // gt boxes per image
#define NA 2064     // all boxes per image
#define NROI 512    // sampled per image
#define CCH 256     // channels
#define HW 64       // feature H = W
#define DPOOL 12544 // 256 * 49
#define FC 1024
#define NHEAD 401   // 81 + 320
#define NHEADP 512  // padded to multiple of BN

// ---- CDNA5 async global->LDS path (guarded; falls back to ds_store) -------
#if defined(__has_builtin)
#  if __has_builtin(__builtin_amdgcn_global_load_async_to_lds_b128) && \
      __has_builtin(__builtin_amdgcn_s_wait_asynccnt)
#    define USE_ASYNC_LDS 1
#  endif
#endif
#ifndef USE_ASYNC_LDS
#  define USE_ASYNC_LDS 0
#endif

#if USE_ASYNC_LDS
// Builtin signature (from hipcc diagnostic): param 0 is `int4 __device__*`,
// i.e. a pointer to int32x4 in address_space(1); LDS side is address_space(3).
typedef int v4i __attribute__((ext_vector_type(4)));
typedef __attribute__((address_space(1))) v4i g_v4i;
typedef __attribute__((address_space(3))) v4i l_v4i;
static __device__ __forceinline__ void async_cp16(const void* g, void* l) {
    // Generic->LDS: low 32 bits of a flat shared pointer are the LDS offset
    // (CDNA5 aperture rule).
    __builtin_amdgcn_global_load_async_to_lds_b128(
        (g_v4i*)(uintptr_t)g,
        (l_v4i*)(uintptr_t)(uint32_t)(uintptr_t)l,
        0, 0);
}
static __device__ __forceinline__ void async_wait0() {
    __builtin_amdgcn_s_wait_asynccnt(0);
}
#else
static __device__ __forceinline__ void async_wait0() {}
#endif

// ---------------------------------------------------------------------------
// 1) IoU match + FG/BG sampling (reproduces argsort(argsort) ranking with
//    stable tie-break by index, emits in top_k order).
// ---------------------------------------------------------------------------
__global__ void sample_rois_kernel(const float* __restrict__ prop,   // [B][NP][4]
                                   const float* __restrict__ gtb,    // [B][NG][4]
                                   const float* __restrict__ rscore, // [B][NA]
                                   float* __restrict__ rois)         // [B*512][4]
{
    const int img = blockIdx.x;
    const int tid = threadIdx.x;

    __shared__ float gs[NG][4];
    __shared__ float rnd[NA];
    __shared__ unsigned char fgm[NA];
    __shared__ unsigned char sel[NA];
    __shared__ int cfg, cbg;

    if (tid == 0) { cfg = 0; cbg = 0; }
    if (tid < NG * 4) ((float*)gs)[tid] = gtb[img * NG * 4 + tid];
    __syncthreads();

    for (int i = tid; i < NA; i += blockDim.x) {
        float bx1, by1, bx2, by2;
        if (i < NP) {
            const float* p = prop + (size_t)(img * NP + i) * 4;
            bx1 = p[0]; by1 = p[1]; bx2 = p[2]; by2 = p[3];
        } else {
            bx1 = gs[i - NP][0]; by1 = gs[i - NP][1];
            bx2 = gs[i - NP][2]; by2 = gs[i - NP][3];
        }
        float ap = (bx2 - bx1) * (by2 - by1);
        float best = 0.f;
        for (int g = 0; g < NG; ++g) {
            float gx1 = gs[g][0], gy1 = gs[g][1], gx2 = gs[g][2], gy2 = gs[g][3];
            float iw = fminf(gx2, bx2) - fmaxf(gx1, bx1);
            float ih = fminf(gy2, by2) - fmaxf(gy1, by1);
            iw = fmaxf(iw, 0.f); ih = fmaxf(ih, 0.f);
            float inter = iw * ih;
            float ag = (gx2 - gx1) * (gy2 - gy1);
            float iou = inter > 0.f ? inter / (ag + ap - inter) : 0.f;
            best = fmaxf(best, iou);
        }
        int fg = (best >= 0.5f) ? 1 : 0;  // gt classes are always < 80
        fgm[i] = (unsigned char)fg;
        rnd[i] = rscore[(size_t)img * NA + i];
        if (fg) atomicAdd(&cfg, 1); else atomicAdd(&cbg, 1);
    }
    __syncthreads();

    const int nfg = min(cfg, 128);
    const int nbg = min(NROI - nfg, cbg);

    for (int i = tid; i < NA; i += blockDim.x) {
        int fg = fgm[i];
        float ri = rnd[i];
        int rank = 0;
        for (int j = 0; j < NA; ++j) {
            if (fgm[j] == fg && (rnd[j] < ri || (rnd[j] == ri && j < i))) rank++;
        }
        sel[i] = fg ? (rank < nfg ? 2 : 0) : (rank < nbg ? 1 : 0);
    }
    __syncthreads();

    if (tid == 0) {
        int pos = 0;
        for (int s = 2; s >= 0 && pos < NROI; --s) {
            for (int i = 0; i < NA && pos < NROI; ++i) {
                if (sel[i] != (unsigned char)s) continue;
                float bx1, by1, bx2, by2;
                if (i < NP) {
                    const float* p = prop + (size_t)(img * NP + i) * 4;
                    bx1 = p[0]; by1 = p[1]; bx2 = p[2]; by2 = p[3];
                } else {
                    bx1 = gs[i - NP][0]; by1 = gs[i - NP][1];
                    bx2 = gs[i - NP][2]; by2 = gs[i - NP][3];
                }
                float* o = rois + (size_t)(img * NROI + pos) * 4;
                o[0] = bx1; o[1] = by1; o[2] = bx2; o[3] = by2;
                pos++;
            }
        }
    }
}

// ---------------------------------------------------------------------------
// 2) NCHW -> NHWC feature transpose.
// ---------------------------------------------------------------------------
__global__ void transpose_feat_kernel(const float* __restrict__ f,  // [B][C][H][W]
                                      float* __restrict__ fT)       // [B][H][W][C]
{
    const int b = blockIdx.x >> 8;
    const int c = blockIdx.x & 255;
    const float* src = f + (size_t)(b * CCH + c) * (HW * HW);
    for (int p = threadIdx.x; p < HW * HW; p += blockDim.x)
        fT[((size_t)b * HW * HW + p) * CCH + c] = src[p];
}

// ---------------------------------------------------------------------------
// 3) ROIAlign -> pooled f16, bin-major layout pooled[r][bin*256 + c].
// ---------------------------------------------------------------------------
__global__ void roi_align_kernel(const float* __restrict__ fT,   // [B][64][64][256]
                                 const float* __restrict__ rois, // [1024][4]
                                 h16_t* __restrict__ pooled)     // [1024][12544]
{
    const int r = blockIdx.x;
    const int c = threadIdx.x;
    const int b = r >> 9; // r / 512
    const float* rb = rois + (size_t)r * 4;
    const float x1 = rb[0] / 16.f - 0.5f, y1 = rb[1] / 16.f - 0.5f;
    const float x2 = rb[2] / 16.f - 0.5f, y2 = rb[3] / 16.f - 0.5f;
    const float sx = (x2 - x1) / 7.f, sy = (y2 - y1) / 7.f;
    const float* fb = fT + (size_t)b * HW * HW * CCH;

    for (int bin = 0; bin < 49; ++bin) {
        const int ph = bin / 7, pw = bin % 7;
        float acc = 0.f;
#pragma unroll
        for (int s = 0; s < 4; ++s) {
            const int iy = ph * 2 + (s >> 1), ix = pw * 2 + (s & 1);
            float xs = x1 + ((float)ix + 0.5f) * 0.5f * sx;
            float ys = y1 + ((float)iy + 0.5f) * 0.5f * sy;
            xs = fminf(fmaxf(xs, 0.f), 63.f);
            ys = fminf(fmaxf(ys, 0.f), 63.f);
            const int x0 = (int)floorf(xs), y0 = (int)floorf(ys);
            const int x1i = min(x0 + 1, 63), y1i = min(y0 + 1, 63);
            const float lx = xs - (float)x0, ly = ys - (float)y0;
            const float v00 = fb[((size_t)(y0  * HW + x0 ) * CCH) + c];
            const float v01 = fb[((size_t)(y0  * HW + x1i) * CCH) + c];
            const float v10 = fb[((size_t)(y1i * HW + x0 ) * CCH) + c];
            const float v11 = fb[((size_t)(y1i * HW + x1i) * CCH) + c];
            acc += v00 * (1.f - ly) * (1.f - lx) + v01 * (1.f - ly) * lx
                 + v10 * ly * (1.f - lx)         + v11 * ly * lx;
        }
        pooled[(size_t)r * DPOOL + bin * CCH + c] = (h16_t)(acc * 0.25f);
    }
}

// ---------------------------------------------------------------------------
// 4) Weight conversions (f32 -> f16, stored N-major Bt[n][k]; fc1 folds in
//    the pooled-layout permutation s=c*49+bin -> k'=bin*256+c).
// ---------------------------------------------------------------------------
__global__ void convert_fc1w_kernel(const float* __restrict__ w,  // [12544][1024]
                                    h16_t* __restrict__ wh)       // [1024][12544]
{
    const int s  = blockIdx.x;
    const int kp = (s % 49) * CCH + (s / 49);
    for (int n = threadIdx.x; n < FC; n += blockDim.x)
        wh[(size_t)n * DPOOL + kp] = (h16_t)w[(size_t)s * FC + n];
}

__global__ void transpose_w_kernel(const float* __restrict__ w,   // [K][N]
                                   h16_t* __restrict__ wh,        // [N][K]
                                   int K, int N)
{
    const int k = blockIdx.x;
    for (int n = threadIdx.x; n < N; n += blockDim.x)
        wh[(size_t)n * K + k] = (h16_t)w[(size_t)k * N + n];
}

__global__ void build_headw_kernel(const float* __restrict__ clsw, // [1024][81]
                                   const float* __restrict__ boxw, // [1024][320]
                                   h16_t* __restrict__ wh)         // [512][1024]
{
    const int k = blockIdx.x;
    const int n = threadIdx.x; // 512
    float v = 0.f;
    if (n < 81)         v = clsw[(size_t)k * 81 + n];
    else if (n < NHEAD) v = boxw[(size_t)k * 320 + (n - 81)];
    wh[(size_t)n * FC + k] = (h16_t)v;
}

__global__ void build_headb_kernel(const float* __restrict__ clsb,
                                   const float* __restrict__ boxb,
                                   float* __restrict__ hb)         // [512]
{
    const int n = threadIdx.x;
    float v = 0.f;
    if (n < 81)         v = clsb[n];
    else if (n < NHEAD) v = boxb[n - 81];
    hb[n] = v;
}

// ---------------------------------------------------------------------------
// 5) Split-K WMMA GEMM: Cp[s][m][n] = sum_{k in slice s} A[m][k] * Bt[n][k].
//    Block tile 128x128, BK=32, 8 waves, wave tile 64x32 (4x2 of 16x16):
//    8 x v_wmma_f32_16x16x32_f16 per K-step vs 12 ds_load_b128.
//    Double-buffered LDS (40 KB); async global->LDS when available.
//    +8-half row pad (80-B rows): fragment-load start banks cover all 64
//    banks disjointly -> conflict-free ds_load_b128.
// ---------------------------------------------------------------------------
#define BM 128
#define BN 128
#define BK 32
#define LDKH 40     // BK + 8 halves
#define SPLITK 4

__global__ __launch_bounds__(256) void wmma_gemm_splitk_kernel(
    const h16_t* __restrict__ A,  int lda,   // [1024][K]
    const h16_t* __restrict__ Bt, int ldb,   // [Npad][K]
    float* __restrict__ Cp,                  // [SPLITK][1024][Npad]
    int Npad, int K)
{
    __shared__ __align__(16) h16_t As[2][BM][LDKH];
    __shared__ __align__(16) h16_t Bs[2][BN][LDKH];

    const int tid  = threadIdx.x;
    const int lane = tid & 31;
    const int w    = tid >> 5;
    const int wm   = w >> 2;          // 0..1 : 64 rows each
    const int wn   = w & 3;           // 0..3 : 32 cols each
    const int row0 = blockIdx.x * BM;
    const int col0 = blockIdx.y * BN;
    const int s    = blockIdx.z;
    const int kbeg = s * (K / SPLITK);
    const int nt   = (K / SPLITK) / BK;

    v8f acc[4][2] = {};

    auto load_tile = [&](int t, int b) {
        const int k0 = kbeg + t * BK;
#pragma unroll
        for (int p = 0; p < 2; ++p) {
            const int ch = tid + p * 256;        // 512 chunks per matrix
            const int rr = ch >> 2, cc = (ch & 3) * 8;
            const h16_t* ga = A  + (size_t)(row0 + rr) * lda + k0 + cc;
            const h16_t* gb = Bt + (size_t)(col0 + rr) * ldb + k0 + cc;
#if USE_ASYNC_LDS
            async_cp16(ga, &As[b][rr][cc]);
            async_cp16(gb, &Bs[b][rr][cc]);
#else
            *(half8*)&As[b][rr][cc] = *(const half8*)ga;
            *(half8*)&Bs[b][rr][cc] = *(const half8*)gb;
#endif
        }
    };

    load_tile(0, 0);
    async_wait0();
    __syncthreads();

    const int ls = lane & 15;
    const int ko = (lane >> 4) * 8;   // K {0..7,16..23} / {8..15,24..31}

    for (int t = 0; t < nt; ++t) {
        const int cur = t & 1;
        if (t + 1 < nt) load_tile(t + 1, cur ^ 1);

        v16h af[4], bf[2];
#pragma unroll
        for (int i = 0; i < 4; ++i) {
            const h16_t* pa = &As[cur][wm * 64 + i * 16 + ls][ko];
            half8 lo = *(const half8*)pa;
            half8 hi = *(const half8*)(pa + 16);
            af[i] = __builtin_shufflevector(lo, hi, 0, 1, 2, 3, 4, 5, 6, 7,
                                            8, 9, 10, 11, 12, 13, 14, 15);
        }
#pragma unroll
        for (int j = 0; j < 2; ++j) {
            const h16_t* pb = &Bs[cur][wn * 32 + j * 16 + ls][ko];
            half8 lo = *(const half8*)pb;
            half8 hi = *(const half8*)(pb + 16);
            bf[j] = __builtin_shufflevector(lo, hi, 0, 1, 2, 3, 4, 5, 6, 7,
                                            8, 9, 10, 11, 12, 13, 14, 15);
        }
#pragma unroll
        for (int i = 0; i < 4; ++i)
#pragma unroll
            for (int j = 0; j < 2; ++j)
                acc[i][j] = __builtin_amdgcn_wmma_f32_16x16x32_f16(
                    false, af[i], false, bf[j], (short)0, acc[i][j], false, false);

        async_wait0();
        __syncthreads();
    }

    // Partial writeout. C layout: VGPR v -> m = v + 8*(lane>>4), n = lane&15.
    const int mo = 8 * (lane >> 4);
    float* base = Cp + (size_t)s * 1024 * Npad;
#pragma unroll
    for (int i = 0; i < 4; ++i) {
#pragma unroll
        for (int j = 0; j < 2; ++j) {
            const int n = col0 + wn * 32 + j * 16 + ls;
#pragma unroll
            for (int v = 0; v < 8; ++v) {
                const int m = row0 + wm * 64 + i * 16 + v + mo;
                base[(size_t)m * Npad + n] = acc[i][j][v];
            }
        }
    }
}

// ---------------------------------------------------------------------------
// 6) Split-K combine + bias + optional ReLU; emits f16 (next layer) and/or
//    f32 (final output, guarded by true N).
// ---------------------------------------------------------------------------
__global__ void combine_kernel(const float* __restrict__ Cp, int Npad,
                               const float* __restrict__ bias,
                               float* __restrict__ Cf, int ldcf,
                               h16_t* __restrict__ Ch, int ldch,
                               int N, int relu)
{
    const int m = blockIdx.x;
    for (int n = threadIdx.x; n < Npad; n += blockDim.x) {
        float v = 0.f;
#pragma unroll
        for (int s = 0; s < SPLITK; ++s)
            v += Cp[((size_t)(s * 1024 + m)) * Npad + n];
        v += bias[n];
        if (relu) v = fmaxf(v, 0.f);
        if (Ch) Ch[(size_t)m * ldch + n] = (h16_t)v;
        if (Cf && n < N) Cf[(size_t)m * ldcf + n] = v;
    }
}

// ---------------------------------------------------------------------------
// Launch
// ---------------------------------------------------------------------------
extern "C" void kernel_launch(void* const* d_in, const int* in_sizes, int n_in,
                              void* d_out, int out_size, void* d_ws, size_t ws_size,
                              hipStream_t stream) {
    const float* feat = (const float*)d_in[0];
    const float* prop = (const float*)d_in[1];
    const float* gtb  = (const float*)d_in[2];
    // d_in[3] gt_classes: unused (classes always < 80 -> fg == matched)
    const float* rsc  = (const float*)d_in[4];
    const float* fc1w = (const float*)d_in[5];
    const float* fc1b = (const float*)d_in[6];
    const float* fc2w = (const float*)d_in[7];
    const float* fc2b = (const float*)d_in[8];
    const float* clsw = (const float*)d_in[9];
    const float* clsb = (const float*)d_in[10];
    const float* boxw = (const float*)d_in[11];
    const float* boxb = (const float*)d_in[12];
    float* out = (float*)d_out;

    char* ws = (char*)d_ws;
    size_t off = 0;
    auto carve = [&](size_t bytes) -> void* {
        void* p = ws + off;
        off = (off + bytes + 255) & ~(size_t)255;
        return p;
    };
    float* rois    = (float*)carve((size_t)1024 * 4 * 4);
    float* fT      = (float*)carve((size_t)2 * HW * HW * CCH * 4);     // 8.4 MB
    h16_t* pooledh = (h16_t*)carve((size_t)1024 * DPOOL * 2);          // 25.7 MB
    h16_t* fc1h    = (h16_t*)carve((size_t)FC * DPOOL * 2);            // 25.7 MB
    h16_t* x1h     = (h16_t*)carve((size_t)1024 * FC * 2);
    h16_t* fc2h    = (h16_t*)carve((size_t)FC * FC * 2);
    h16_t* x2h     = (h16_t*)carve((size_t)1024 * FC * 2);
    h16_t* headh   = (h16_t*)carve((size_t)NHEADP * FC * 2);
    float* headb   = (float*)carve((size_t)NHEADP * 4);
    float* Cp      = (float*)carve((size_t)SPLITK * 1024 * FC * 4);    // 16.8 MB

    sample_rois_kernel<<<2, 256, 0, stream>>>(prop, gtb, rsc, rois);
    transpose_feat_kernel<<<512, 256, 0, stream>>>(feat, fT);
    convert_fc1w_kernel<<<DPOOL, 256, 0, stream>>>(fc1w, fc1h);
    transpose_w_kernel<<<FC, 256, 0, stream>>>(fc2w, fc2h, FC, FC);
    build_headw_kernel<<<FC, NHEADP, 0, stream>>>(clsw, boxw, headh);
    build_headb_kernel<<<1, NHEADP, 0, stream>>>(clsb, boxb, headb);
    roi_align_kernel<<<1024, 256, 0, stream>>>(fT, rois, pooledh);

    // FC1: 1024 x 1024 x 12544
    dim3 g1(1024 / BM, FC / BN, SPLITK);
    wmma_gemm_splitk_kernel<<<g1, 256, 0, stream>>>(pooledh, DPOOL, fc1h, DPOOL,
                                                    Cp, FC, DPOOL);
    combine_kernel<<<1024, 256, 0, stream>>>(Cp, FC, fc1b, nullptr, 0,
                                             x1h, FC, FC, 1);
    // FC2: 1024 x 1024 x 1024
    wmma_gemm_splitk_kernel<<<g1, 256, 0, stream>>>(x1h, FC, fc2h, FC,
                                                    Cp, FC, FC);
    combine_kernel<<<1024, 256, 0, stream>>>(Cp, FC, fc2b, nullptr, 0,
                                             x2h, FC, FC, 1);
    // Heads: 1024 x 512(pad of 401) x 1024
    dim3 g3(1024 / BM, NHEADP / BN, SPLITK);
    wmma_gemm_splitk_kernel<<<g3, 256, 0, stream>>>(x2h, FC, headh, FC,
                                                    Cp, NHEADP, FC);
    combine_kernel<<<1024, 256, 0, stream>>>(Cp, NHEADP, headb, out, NHEAD,
                                             nullptr, 0, NHEAD, 0);
}
